// FNO1d_59141699666054
// MI455X (gfx1250) — matrix operand order repaired and checked by reading, not compiled, not run on hardware
//
#include <hip/hip_runtime.h>
#include <hip/hip_bf16.h>

// FNO1d on gfx1250: truncated-DFT-as-GEMM formulation, all heavy math on
// v_wmma_f32_16x16x32_bf16 with f32 accumulation. ~36 GFLOP, ~1.3 GB traffic
// -> memory-bound at 23.3 TB/s; bf16 WMMA keeps compute far below that.

typedef __bf16 bf16_t;
typedef __attribute__((ext_vector_type(16))) __bf16 v16bf;
typedef __attribute__((ext_vector_type(8)))  __bf16 v8bf;
typedef __attribute__((ext_vector_type(8)))  float  v8f;

constexpr int NB = 64, NS = 8192, NW = 64, NM = 16, NL = 4;
constexpr float TWOPI = 6.283185307179586f;

__device__ __forceinline__ v8f wmma_bf16(v16bf a, v16bf b, v8f c) {
  return __builtin_amdgcn_wmma_f32_16x16x32_bf16(false, a, false, b, (short)0, c,
                                                 false, false);
}

// A-operand (16x32 bf16) per-lane gather: two contiguous 8xbf16 (16B) chunks.
__device__ __forceinline__ v16bf load_a16(const bf16_t* p0, const bf16_t* p1) {
  v8bf lo = *(const v8bf*)p0;
  v8bf hi = *(const v8bf*)p1;
  v16bf r;
#pragma unroll
  for (int i = 0; i < 8; ++i) { r[i] = lo[i]; r[8 + i] = hi[i]; }
  return r;
}

// ---------------- setup: trig bases pre-swizzled into WMMA B-operand layout ---
// FBsw: [256 kchunks][2 ngrp][32 lanes][16]  forward basis F[s][j] (1/S folded)
// GBsw: [512 stiles][32 lanes][16]           inverse basis G[j][s] (c_k folded)
__global__ void k_basis(bf16_t* __restrict__ FBsw, bf16_t* __restrict__ GBsw) {
  int idx = blockIdx.x * blockDim.x + threadIdx.x;
  if (idx < 256 * 2 * 32 * 16) {
    int p = idx & 15, lane = (idx >> 4) & 31, ng = (idx >> 9) & 1, kc = idx >> 10;
    int j = ng * 16 + (lane & 15);
    int s = kc * 32 + ((lane < 16) ? 0 : 8) + ((p < 8) ? p : p + 8);
    int k = j >> 1;
    float th = TWOPI * (float)((k * s) % NS) / (float)NS;
    float v = (j & 1) ? (-sinf(th) / (float)NS) : (cosf(th) / (float)NS);
    FBsw[idx] = (bf16_t)v;
  } else {
    int i = idx - 256 * 2 * 32 * 16;
    if (i >= 512 * 32 * 16) return;
    int p = i & 15, lane = (i >> 4) & 31, st = i >> 9;
    int j = ((lane < 16) ? 0 : 8) + ((p < 8) ? p : p + 8);
    int s = st * 16 + (lane & 15);
    int k = j >> 1;
    float c = (k == 0) ? 1.0f : 2.0f;
    float th = TWOPI * (float)((k * s) % NS) / (float)NS;
    float v = (j & 1) ? (-c * sinf(th)) : (c * cosf(th));
    GBsw[i] = (bf16_t)v;
  }
}

// convA: conv weights in A-operand layout [L][4 ogrp][2 ktile][32][16]
// qBsw : q_w in B-operand layout [2 ktile][2 ngrp][32][16]
__global__ void k_wswz(const float* __restrict__ conv_w, const float* __restrict__ q_w,
                       bf16_t* __restrict__ convA, bf16_t* __restrict__ qBsw) {
  int i = blockIdx.x * blockDim.x + threadIdx.x;
  if (i < NL * 4 * 2 * 32 * 16) {
    int p = i & 15, lane = (i >> 4) & 31, kt = (i >> 9) & 1, og = (i >> 10) & 3,
        l = i >> 12;
    int row = og * 16 + (lane & 15);
    int k = kt * 32 + ((lane < 16) ? 0 : 16) + p;
    convA[i] = (bf16_t)conv_w[(l * NW + row) * NW + k];
  } else {
    int i2 = i - NL * 4 * 2 * 32 * 16;
    if (i2 >= 2 * 2 * 32 * 16) return;
    int p = i2 & 15, lane = (i2 >> 4) & 31, ng = (i2 >> 9) & 1, kt = i2 >> 10;
    int k = kt * 32 + ((lane < 16) ? 0 : 8) + ((p < 8) ? p : p + 8);
    int n = ng * 16 + (lane & 15);
    qBsw[i2] = (bf16_t)q_w[k * 32 + n];
  }
}

// ---------------- lift: x@(2xW) -> h in both bf16 layouts ---------------------
__global__ __launch_bounds__(256) void k_lift(const float* __restrict__ x,
    const float* __restrict__ p_w, const float* __restrict__ p_b,
    bf16_t* __restrict__ hb, bf16_t* __restrict__ hbT) {
  __shared__ float xs[64][2];
  int b = blockIdx.x, s0 = blockIdx.y * 64;
  int t = threadIdx.x;
  if (t < 128) xs[t >> 1][t & 1] = x[((size_t)b * NS + s0 + (t >> 1)) * 2 + (t & 1)];
  __syncthreads();
  {  // (B,W,S) layout: thread owns (w, 16 contiguous s)
    int w = t >> 2, sc = t & 3;
    float pw0 = p_w[w], pw1 = p_w[NW + w], pb = p_b[w];
    v16bf v;
#pragma unroll
    for (int i = 0; i < 16; ++i) {
      float h = xs[sc * 16 + i][0] * pw0 + xs[sc * 16 + i][1] * pw1 + pb;
      v[i] = (bf16_t)h;
    }
    *(v16bf*)(hb + ((size_t)(b * NW + w)) * NS + s0 + sc * 16) = v;
  }
  {  // (B,S,W) layout: thread owns (s, 16 contiguous w)
    int sl = t >> 2, wc = (t & 3) * 16;
    float x0 = xs[sl][0], x1 = xs[sl][1];
    v16bf v;
#pragma unroll
    for (int i = 0; i < 16; ++i) {
      int ww = wc + i;
      float h = x0 * p_w[ww] + x1 * p_w[NW + ww] + p_b[ww];
      v[i] = (bf16_t)h;
    }
    *(v16bf*)(hbT + ((size_t)b * NS + s0 + sl) * NW + wc) = v;
  }
}

// ---------------- forward truncated DFT: (4096 x 8192) @ (8192 x 32) ---------
// K split across 16 slices (partials reduced deterministically by k_ftred).
__global__ __launch_bounds__(256) void k_dft(const bf16_t* __restrict__ hb,
    const bf16_t* __restrict__ FBsw, float* __restrict__ FTpart) {
  int mt = blockIdx.x;
  int wv = threadIdx.x >> 5, lane = threadIdx.x & 31;
  int kslice = blockIdx.y * 8 + wv;
  int r = mt * 16 + (lane & 15);
  const bf16_t* arow = hb + (size_t)r * NS;
  int klo = (lane < 16) ? 0 : 8;
  v8f acc0 = {}, acc1 = {};
  for (int c = 0; c < 16; ++c) {
    int k0 = kslice * 512 + c * 32;
    __builtin_prefetch(arow + k0 + 128, 0, 3);
    v16bf a = load_a16(arow + k0 + klo, arow + k0 + 16 + klo);
    const bf16_t* fb = FBsw + (size_t)(k0 >> 5) * 1024;
    v16bf b0 = *(const v16bf*)(fb + lane * 16);
    v16bf b1 = *(const v16bf*)(fb + 512 + lane * 16);
    acc0 = wmma_bf16(a, b0, acc0);
    acc1 = wmma_bf16(a, b1, acc1);
  }
  int moff = (lane < 16) ? 0 : 8;
  int col = lane & 15;
  float* outp = FTpart + (size_t)kslice * (4096 * 32);
#pragma unroll
  for (int i = 0; i < 8; ++i) {
    outp[(size_t)(mt * 16 + moff + i) * 32 + col] = acc0[i];
    outp[(size_t)(mt * 16 + moff + i) * 32 + 16 + col] = acc1[i];
  }
}

__global__ void k_ftred(const float* __restrict__ FTpart, float* __restrict__ FT) {
  int i = blockIdx.x * blockDim.x + threadIdx.x;
  if (i >= 4096 * 32) return;
  float s = 0.f;
#pragma unroll
  for (int k = 0; k < 16; ++k) s += FTpart[(size_t)k * (4096 * 32) + i];
  FT[i] = s;
}

// ---------------- complex mode mix; writes OFT directly in A-operand layout --
__global__ __launch_bounds__(64) void k_mix(const float* __restrict__ FT,
    const float* __restrict__ swr, const float* __restrict__ swi,
    bf16_t* __restrict__ OFTA, int l) {
  int b = blockIdx.x / NM, k = blockIdx.x % NM;
  int o = threadIdx.x;
  __shared__ float fr[NW], fi[NW];
  fr[o] = FT[(size_t)(b * NW + o) * 32 + 2 * k];
  fi[o] = FT[(size_t)(b * NW + o) * 32 + 2 * k + 1];
  __syncthreads();
  float ar = 0.f, ai = 0.f;
  for (int i = 0; i < NW; ++i) {
    float wr = swr[((size_t)(l * NW + i) * NW + o) * NM + k];
    float wi = swi[((size_t)(l * NW + i) * NW + o) * NM + k];
    ar += fr[i] * wr - fi[i] * wi;
    ai += fr[i] * wi + fi[i] * wr;
  }
  int og = o >> 4, m = o & 15;
  int j = 2 * k;
  int g = j >> 3;
  int lane = (g & 1) ? (m + 16) : m;
  int pos = (j & 7) + ((g >> 1) ? 8 : 0);
  size_t idx = ((size_t)(b * 4 + og) * 32 + lane) * 16 + pos;
  OFTA[idx] = (bf16_t)ar;
  OFTA[idx + 1] = (bf16_t)ai;
}

// ---------------- fused iDFT + 1x1 conv + bias + tanh ------------------------
// per wave: 16 o-rows x 16 s-cols tile = 2 conv WMMAs (K=64) + 1 iDFT WMMA (K=32)
__global__ __launch_bounds__(256) void k_layer(const bf16_t* __restrict__ hbT_in,
    const bf16_t* __restrict__ OFTA, const bf16_t* __restrict__ GBsw,
    const bf16_t* __restrict__ convA_l, const float* __restrict__ conv_b_l,
    bf16_t* __restrict__ hb_out, bf16_t* __restrict__ hbT_out) {
  int b = blockIdx.x, s0 = blockIdx.y * 64;
  int wv = threadIdx.x >> 5, lane = threadIdx.x & 31;
  int og = wv & 3, sh = wv >> 2;
  int o0 = og * 16;
  v16bf aC0 = *(const v16bf*)(convA_l + ((size_t)(og * 2 + 0) * 32 + lane) * 16);
  v16bf aC1 = *(const v16bf*)(convA_l + ((size_t)(og * 2 + 1) * 32 + lane) * 16);
  v16bf aF  = *(const v16bf*)(OFTA + ((size_t)(b * 4 + og) * 32 + lane) * 16);
  int moff = (lane < 16) ? 0 : 8;
  float bias[8];
#pragma unroll
  for (int i = 0; i < 8; ++i) bias[i] = conv_b_l[o0 + moff + i];
  int kb = (lane < 16) ? 0 : 16;
  int sc = lane & 15;
#pragma unroll
  for (int nt = 0; nt < 2; ++nt) {
    int s = s0 + sh * 32 + nt * 16;
    v16bf bG = *(const v16bf*)(GBsw + ((size_t)(s >> 4) * 32 + lane) * 16);
    const bf16_t* hrow = hbT_in + ((size_t)b * NS + s + sc) * NW;
    v16bf bH0 = *(const v16bf*)(hrow + kb);        // conv K = 0..31
    v16bf bH1 = *(const v16bf*)(hrow + 32 + kb);   // conv K = 32..63
    v8f acc = {};
    acc = wmma_bf16(aC0, bH0, acc);
    acc = wmma_bf16(aC1, bH1, acc);
    acc = wmma_bf16(aF, bG, acc);
    v8bf pk;
#pragma unroll
    for (int i = 0; i < 8; ++i) {
      float rv = tanhf(acc[i] + bias[i]);
      pk[i] = (bf16_t)rv;
      hb_out[(size_t)(b * NW + o0 + moff + i) * NS + s + sc] = (bf16_t)rv;
    }
    // transposed layout store: 8 contiguous bf16 per lane
    *(v8bf*)(hbT_out + ((size_t)b * NS + s + sc) * NW + o0 + moff) = pk;
  }
}

// ---------------- head: tanh(h@q_w+q_b)@o_w+o_b ------------------------------
__global__ __launch_bounds__(256) void k_head(const bf16_t* __restrict__ hbT,
    const bf16_t* __restrict__ qBsw, const float* __restrict__ q_b,
    const float* __restrict__ o_w, const float* __restrict__ o_b,
    float* __restrict__ out) {
  int tile = blockIdx.x * 8 + (threadIdx.x >> 5);
  int lane = threadIdx.x & 31;
  int b = tile >> 9;            // 512 s-tiles per batch
  int s0 = (tile & 511) * 16;
  const bf16_t* arow = hbT + ((size_t)b * NS + s0 + (lane & 15)) * NW;
  int klo = (lane < 16) ? 0 : 8;
  v16bf a0 = load_a16(arow + klo, arow + 16 + klo);
  v16bf a1 = load_a16(arow + 32 + klo, arow + 48 + klo);
  v16bf b00 = *(const v16bf*)(qBsw + ((size_t)(0 * 2 + 0) * 32 + lane) * 16);
  v16bf b01 = *(const v16bf*)(qBsw + ((size_t)(0 * 2 + 1) * 32 + lane) * 16);
  v16bf b10 = *(const v16bf*)(qBsw + ((size_t)(1 * 2 + 0) * 32 + lane) * 16);
  v16bf b11 = *(const v16bf*)(qBsw + ((size_t)(1 * 2 + 1) * 32 + lane) * 16);
  v8f c0 = {}, c1 = {};
  c0 = wmma_bf16(a0, b00, c0);
  c0 = wmma_bf16(a1, b10, c0);
  c1 = wmma_bf16(a0, b01, c1);
  c1 = wmma_bf16(a1, b11, c1);
  int n = lane & 15;
  float qb0 = q_b[n], qb1 = q_b[16 + n];
  float ow0 = o_w[n], ow1 = o_w[16 + n];
  float p[8];
#pragma unroll
  for (int i = 0; i < 8; ++i)
    p[i] = tanhf(c0[i] + qb0) * ow0 + tanhf(c1[i] + qb1) * ow1;
#pragma unroll
  for (int m = 1; m <= 8; m <<= 1) {
#pragma unroll
    for (int i = 0; i < 8; ++i) p[i] += __shfl_xor(p[i], m, 32);
  }
  if ((lane & 15) == 0) {
    int moff = (lane < 16) ? 0 : 8;
    float ob = o_b[0];
#pragma unroll
    for (int i = 0; i < 8; ++i) out[(size_t)b * NS + s0 + moff + i] = p[i] + ob;
  }
}

extern "C" void kernel_launch(void* const* d_in, const int* in_sizes, int n_in,
                              void* d_out, int out_size, void* d_ws, size_t ws_size,
                              hipStream_t stream) {
  (void)in_sizes; (void)n_in; (void)out_size; (void)ws_size;
  const float* x     = (const float*)d_in[0];
  const float* p_w   = (const float*)d_in[1];
  const float* p_b   = (const float*)d_in[2];
  const float* swr   = (const float*)d_in[3];
  const float* swi   = (const float*)d_in[4];
  const float* convw = (const float*)d_in[5];
  const float* convb = (const float*)d_in[6];
  const float* q_w   = (const float*)d_in[7];
  const float* q_b   = (const float*)d_in[8];
  const float* o_w   = (const float*)d_in[9];
  const float* o_b   = (const float*)d_in[10];

  char* wsp = (char*)d_ws;
  auto take = [&](size_t bytes) {
    char* p = wsp; wsp += (bytes + 255) & ~(size_t)255; return p;
  };
  size_t hbytes = (size_t)NB * NW * NS * sizeof(bf16_t);     // 64 MB each
  bf16_t* hb0  = (bf16_t*)take(hbytes);
  bf16_t* hbT0 = (bf16_t*)take(hbytes);
  bf16_t* hb1  = (bf16_t*)take(hbytes);
  bf16_t* hbT1 = (bf16_t*)take(hbytes);
  float*  FTpart = (float*)take((size_t)16 * 4096 * 32 * sizeof(float));
  float*  FT     = (float*)take((size_t)4096 * 32 * sizeof(float));
  bf16_t* OFTA   = (bf16_t*)take((size_t)NB * 4 * 32 * 16 * sizeof(bf16_t));
  bf16_t* FBsw   = (bf16_t*)take((size_t)256 * 2 * 32 * 16 * sizeof(bf16_t));
  bf16_t* GBsw   = (bf16_t*)take((size_t)512 * 32 * 16 * sizeof(bf16_t));
  bf16_t* convA  = (bf16_t*)take((size_t)NL * 4 * 2 * 32 * 16 * sizeof(bf16_t));
  bf16_t* qBsw   = (bf16_t*)take((size_t)2 * 2 * 32 * 16 * sizeof(bf16_t));

  bf16_t* hb[2]  = { hb0, hb1 };
  bf16_t* hbT[2] = { hbT0, hbT1 };

  k_basis<<<524288 / 256, 256, 0, stream>>>(FBsw, GBsw);
  k_wswz<<<(16384 + 2048 + 255) / 256, 256, 0, stream>>>(convw, q_w, convA, qBsw);
  k_lift<<<dim3(NB, NS / 64), 256, 0, stream>>>(x, p_w, p_b, hb[0], hbT[0]);
  int cur = 0;
  for (int l = 0; l < NL; ++l) {
    k_dft<<<dim3(256, 2), 256, 0, stream>>>(hb[cur], FBsw, FTpart);
    k_ftred<<<(4096 * 32) / 256, 256, 0, stream>>>(FTpart, FT);
    k_mix<<<NB * NM, 64, 0, stream>>>(FT, swr, swi, OFTA, l);
    k_layer<<<dim3(NB, NS / 64), 256, 0, stream>>>(hbT[cur], OFTA, GBsw,
        convA + (size_t)l * 4096, convb + l * NW, hb[1 - cur], hbT[1 - cur]);
    cur ^= 1;
  }
  k_head<<<(NB * NS / 16) / 8, 256, 0, stream>>>(hbT[cur], qBsw, q_b, o_w, o_b,
                                                 (float*)d_out);
}
